// GNNEncoder_29850022707388
// MI455X (gfx1250) — matrix operand order repaired and checked by reading, not compiled, not run on hardware
//
#include <hip/hip_runtime.h>
#include <hip/hip_bf16.h>

// ---------------------------------------------------------------------------
// GNN encoder forward for MI455X (gfx1250, wave32, WMMA).
// H=128, B=2, N=256, L=3.  All H x H linears run on v_wmma_f32_16x16x32_f16,
// with weights staged to LDS via the gfx1250 async-to-LDS path when available.
// ---------------------------------------------------------------------------

typedef __attribute__((ext_vector_type(16))) _Float16 v16h;
typedef __attribute__((ext_vector_type(8)))  _Float16 v8h;
typedef __attribute__((ext_vector_type(8)))  float    v8f;
typedef __attribute__((ext_vector_type(4)))  int      v4i;

#define HD        128
#define NPT       256
#define NBAT      2
#define NLAYER    3
#define M_EDGE    (NBAT*NPT*NPT)    /* 131072 */
#define M_NODE    (NBAT*NPT)        /* 512    */
#define LOG_TEMP  9.210340371976184f
#define LN_EPS    1e-5f
#define WROW      136               /* LDS row pitch in halves (16B pad) */

#if defined(__has_builtin)
# if __has_builtin(__builtin_amdgcn_global_load_async_to_lds_b128) && \
     __has_builtin(__builtin_amdgcn_s_wait_asynccnt)
#  define USE_ASYNC_LDS 1
# endif
#endif

// ------------------------------ reductions ---------------------------------
__device__ __forceinline__ float bred_sum(float v, float* sb, int tid, int n) {
    sb[tid] = v; __syncthreads();
    for (int s = n >> 1; s > 0; s >>= 1) {
        if (tid < s) sb[tid] += sb[tid + s];
        __syncthreads();
    }
    float r = sb[0]; __syncthreads(); return r;
}
__device__ __forceinline__ float bred_max(float v, float* sb, int tid, int n) {
    sb[tid] = v; __syncthreads();
    for (int s = n >> 1; s > 0; s >>= 1) {
        if (tid < s) sb[tid] = fmaxf(sb[tid], sb[tid + s]);
        __syncthreads();
    }
    float r = sb[0]; __syncthreads(); return r;
}
__device__ __forceinline__ float bred_min(float v, float* sb, int tid, int n) {
    sb[tid] = v; __syncthreads();
    for (int s = n >> 1; s > 0; s >>= 1) {
        if (tid < s) sb[tid] = fminf(sb[tid], sb[tid + s]);
        __syncthreads();
    }
    float r = sb[0]; __syncthreads(); return r;
}

// --------------------------- weight conversion -----------------------------
__global__ void k_cvt_f16(const float* __restrict__ src, _Float16* __restrict__ dst, int n) {
    int i = blockIdx.x * blockDim.x + threadIdx.x;
    if (i < n) dst[i] = (_Float16)src[i];
}

// ------------------- stage 128x128 f16 weights into LDS --------------------
// Global layout: [n][k] row-major, 128 halves/row.  LDS layout: padded rows
// of WROW halves so B-fragment reads are bank-conflict free.
__device__ __forceinline__ void stage_weights(const _Float16* __restrict__ Wh,
                                              _Float16* WS) {
    for (int c = threadIdx.x; c < 128 * 16; c += 256) {
        int row = c >> 4, seg = c & 15;           // 16B segments per row
        _Float16* dst = WS + row * WROW + seg * 8;
        const _Float16* src = Wh + row * HD + seg * 8;
#ifdef USE_ASYNC_LDS
        __builtin_amdgcn_global_load_async_to_lds_b128((v4i*)src, (v4i*)dst, 0, 0);
#else
        *(uint4*)dst = *(const uint4*)src;
#endif
    }
#ifdef USE_ASYNC_LDS
    __builtin_amdgcn_s_wait_asynccnt(0);
#endif
    __syncthreads();
}

// --------------------------- WMMA fragment loads ---------------------------
// A fragment (16x32 f16, CDNA5 layout):
//   lanes 0-15 : M=lane,    K in [kb..kb+7] U [kb+16..kb+23], kb = kt*32
//   lanes 16-31: M=lane-16, K in [kb+8..kb+15] U [kb+24..kb+31]
__device__ __forceinline__ v16h load_a_frag(const float* __restrict__ A,
                                            int m0, int kt, int lane) {
    int row = m0 + (lane & 15);
    int kb  = kt * 32 + ((lane & 16) ? 8 : 0);
    const float* p = A + (size_t)row * HD + kb;
    float4 c0 = *(const float4*)(p);
    float4 c1 = *(const float4*)(p + 4);
    float4 c2 = *(const float4*)(p + 16);
    float4 c3 = *(const float4*)(p + 20);
    v16h a;
    a[0]=(_Float16)c0.x; a[1]=(_Float16)c0.y; a[2]=(_Float16)c0.z; a[3]=(_Float16)c0.w;
    a[4]=(_Float16)c1.x; a[5]=(_Float16)c1.y; a[6]=(_Float16)c1.z; a[7]=(_Float16)c1.w;
    a[8]=(_Float16)c2.x; a[9]=(_Float16)c2.y; a[10]=(_Float16)c2.z; a[11]=(_Float16)c2.w;
    a[12]=(_Float16)c3.x; a[13]=(_Float16)c3.y; a[14]=(_Float16)c3.z; a[15]=(_Float16)c3.w;
    return a;
}

// B fragment (32x16 f16): B[k][n] = W[n][k], read from padded LDS copy.
__device__ __forceinline__ v16h lds_b_frag(const _Float16* WS,
                                           int nt, int kt, int lane) {
    int n  = nt * 16 + (lane & 15);
    int k0 = kt * 32 + ((lane & 16) ? 8 : 0);
    const _Float16* p = WS + n * WROW + k0;       // 16B aligned
    v8h lo = *(const v8h*)p;
    v8h hi = *(const v8h*)(p + 16);
    return __builtin_shufflevector(lo, hi, 0,1,2,3,4,5,6,7,8,9,10,11,12,13,14,15);
}

// --------------------------- generic WMMA GEMM -----------------------------
// Y[m,n] = sum_k A[m,k]*W[n,k] + bias[n]  (+ fused epilogue per MODE)
// MODE 0: plain.  MODE 1: + Ah[b,j,n] + Bh[b,i,n] (m = (b*256+i)*256+j).
// MODE 2: + res[m,n] (residual, may alias Y).
template<int MODE>
__global__ __launch_bounds__(256) void k_gemm(const float* __restrict__ A,
                                              const _Float16* __restrict__ Wh,
                                              const float* __restrict__ bias,
                                              const float* __restrict__ res,
                                              const float* __restrict__ AhP,
                                              const float* __restrict__ BhP,
                                              float* __restrict__ Y, int M) {
    __shared__ _Float16 WS[128 * WROW];           // 34 KB
    int lane = threadIdx.x & 31;
    int wave = blockIdx.x * (blockDim.x >> 5) + (threadIdx.x >> 5);
    int m0 = wave << 4;

    if (m0 < M)   // warm the per-wave activation stream (global_prefetch_b8)
        __builtin_prefetch((const void*)(A + (size_t)(m0 + (lane & 15)) * HD), 0, 0);

    stage_weights(Wh, WS);                        // all waves participate
    if (m0 >= M) return;                          // wave-uniform after barrier

    v8f zero = {0.f,0.f,0.f,0.f,0.f,0.f,0.f,0.f};
    v8f acc[8];
#pragma unroll
    for (int nt = 0; nt < 8; ++nt) acc[nt] = zero;

#pragma unroll
    for (int kt = 0; kt < 4; ++kt) {
        v16h a = load_a_frag(A, m0, kt, lane);
#pragma unroll
        for (int nt = 0; nt < 8; ++nt) {
            v16h b = lds_b_frag(WS, nt, kt, lane);
            acc[nt] = __builtin_amdgcn_wmma_f32_16x16x32_f16(
                false, a, false, b, (short)0, acc[nt], false, false);
        }
    }

    int rbase = m0 + ((lane & 16) ? 8 : 0);
#pragma unroll
    for (int nt = 0; nt < 8; ++nt) {
        int n = nt * 16 + (lane & 15);
        float bv = bias[n];
#pragma unroll
        for (int r = 0; r < 8; ++r) {
            int row = rbase + r;
            float v = acc[nt][r] + bv;
            if (MODE == 1) {
                int j  = row & 255;
                int bi = row >> 8;          // b*256 + i
                int b  = row >> 16;
                v += AhP[((size_t)((b << 8) | j)) * HD + n]
                   + BhP[(size_t)bi * HD + n];
            }
            if (MODE == 2) {
                v += res[(size_t)row * HD + n];
            }
            Y[(size_t)row * HD + n] = v;
        }
    }
}

// ---------------- sine-embedding GEMM (fused A generation) -----------------
// e = (sine(cm) + sine(xt)) @ ee_w^T + 2*ee_b   -- no intermediate storage.
__device__ __forceinline__ v16h sine_frag(float v1, float v2, int kt, int lane) {
    int kb = kt * 32 + ((lane & 16) ? 8 : 0);
    v16h a;
#pragma unroll
    for (int i = 0; i < 16; ++i) {
        int k = kb + (i < 8 ? i : i + 8);
        float inv_dt = __expf(-LOG_TEMP * (float)((k >> 1) << 1) * (1.0f / 128.0f));
        float e1 = v1 * inv_dt, e2 = v2 * inv_dt;
        float s = (k & 1) ? (__cosf(e1) + __cosf(e2)) : (__sinf(e1) + __sinf(e2));
        a[i] = (_Float16)s;
    }
    return a;
}

__global__ __launch_bounds__(256) void k_gemm_sine(const float* __restrict__ cm,
                                                   const int* __restrict__ xt,
                                                   const _Float16* __restrict__ Wh,
                                                   const float* __restrict__ bias,
                                                   float* __restrict__ Y, int M) {
    __shared__ _Float16 WS[128 * WROW];
    int lane = threadIdx.x & 31;
    int wave = blockIdx.x * (blockDim.x >> 5) + (threadIdx.x >> 5);
    int m0 = wave << 4;

    stage_weights(Wh, WS);
    if (m0 >= M) return;

    int row = m0 + (lane & 15);
    float v1 = cm[row];
    float v2 = (float)xt[row];

    v8f zero = {0.f,0.f,0.f,0.f,0.f,0.f,0.f,0.f};
    v8f acc[8];
#pragma unroll
    for (int nt = 0; nt < 8; ++nt) acc[nt] = zero;

#pragma unroll
    for (int kt = 0; kt < 4; ++kt) {
        v16h a = sine_frag(v1, v2, kt, lane);
#pragma unroll
        for (int nt = 0; nt < 8; ++nt) {
            v16h b = lds_b_frag(WS, nt, kt, lane);
            acc[nt] = __builtin_amdgcn_wmma_f32_16x16x32_f16(
                false, a, false, b, (short)0, acc[nt], false, false);
        }
    }

    int rbase = m0 + ((lane & 16) ? 8 : 0);
#pragma unroll
    for (int nt = 0; nt < 8; ++nt) {
        int n = nt * 16 + (lane & 15);
        float bv = 2.0f * bias[n];          // linear applied twice, same bias
#pragma unroll
        for (int r = 0; r < 8; ++r) {
            int r2 = rbase + r;
            Y[(size_t)r2 * HD + n] = acc[nt][r] + bv;
        }
    }
}

// ------------------------- node feature statistics -------------------------
__global__ __launch_bounds__(256) void k_node_stats(const float* __restrict__ cm,
                                                    float* __restrict__ nf) {
    __shared__ float sb[256];
    int bi = blockIdx.x;                 // b*256 + i
    int b = bi >> 8, i = bi & 255;
    int j = threadIdx.x;
    const float* base = cm + ((size_t)b << 16);
    float rv = base[i * 256 + j];
    float cv = base[j * 256 + i];
    float rs = bred_sum(rv, sb, j, 256);
    float rq = bred_sum(rv * rv, sb, j, 256);
    float rmx = bred_max(rv, sb, j, 256);
    float rmn = bred_min(rv, sb, j, 256);
    float cs = bred_sum(cv, sb, j, 256);
    float cq = bred_sum(cv * cv, sb, j, 256);
    float cmx = bred_max(cv, sb, j, 256);
    float cmn = bred_min(cv, sb, j, 256);
    if (j == 0) {
        float mr = rs / 256.f;
        float sr = sqrtf(fmaxf(0.f, (rq - 256.f * mr * mr) / 255.f));
        float mc = cs / 256.f;
        float sc = sqrtf(fmaxf(0.f, (cq - 256.f * mc * mc) / 255.f));
        float* o = nf + (size_t)bi * 8;
        o[0]=mr; o[1]=sr; o[2]=rmx; o[3]=rmn; o[4]=mc; o[5]=sc; o[6]=cmx; o[7]=cmn;
    }
}

// ---------------------------- node feature MLP -----------------------------
__global__ void k_nf1(const float* __restrict__ nf, const float* __restrict__ w1,
                      const float* __restrict__ b1, float* __restrict__ hid) {
    int idx = blockIdx.x * blockDim.x + threadIdx.x;  // bi*256 + o
    if (idx >= M_NODE * 256) return;
    int o = idx & 255, bi = idx >> 8;
    const float* f = nf + (size_t)bi * 8;
    float s = b1[o];
#pragma unroll
    for (int k = 0; k < 8; ++k) s += f[k] * w1[o * 8 + k];
    hid[idx] = fmaxf(s, 0.f);
}

__global__ void k_nf2(const float* __restrict__ hid, const float* __restrict__ w2,
                      const float* __restrict__ b2, float* __restrict__ x) {
    int idx = blockIdx.x * blockDim.x + threadIdx.x;  // bi*128 + h
    if (idx >= M_NODE * HD) return;
    int h = idx & 127, bi = idx >> 7;
    const float* f = hid + (size_t)bi * 256;
    float s = b2[h];
    for (int k = 0; k < 256; ++k) s += f[k] * w2[h * 256 + k];
    x[idx] = s;
}

// -------------------- timestep embedding + per-layer tl --------------------
__global__ __launch_bounds__(128) void k_temb(const int* __restrict__ ts,
        const float* __restrict__ w1, const float* __restrict__ b1,
        const float* __restrict__ w2, const float* __restrict__ b2,
        const float* __restrict__ tw0, const float* __restrict__ tb0,
        const float* __restrict__ tw1, const float* __restrict__ tb1,
        const float* __restrict__ tw2, const float* __restrict__ tb2,
        float* __restrict__ tl_all) {
    __shared__ float traw[128], t1[64], t2r[64];
    int b = blockIdx.x, h = threadIdx.x;
    float t = (float)ts[b];
    if (h < 64) traw[h] = __cosf(t * __expf(-LOG_TEMP * (float)h / 64.f));
    else        traw[h] = __sinf(t * __expf(-LOG_TEMP * (float)(h - 64) / 64.f));
    __syncthreads();
    if (h < 64) {
        float s = b1[h];
        for (int k = 0; k < 128; ++k) s += traw[k] * w1[h * 128 + k];
        t1[h] = fmaxf(s, 0.f);
    }
    __syncthreads();
    if (h < 64) {
        float s = b2[h];
        for (int k = 0; k < 64; ++k) s += t1[k] * w2[h * 64 + k];
        t2r[h] = fmaxf(s, 0.f);            // relu(temb) used by tl layers
    }
    __syncthreads();
    const float* tws[NLAYER] = {tw0, tw1, tw2};
    const float* tbs[NLAYER] = {tb0, tb1, tb2};
    for (int l = 0; l < NLAYER; ++l) {
        float s = tbs[l][h];
        for (int k = 0; k < 64; ++k) s += t2r[k] * tws[l][h * 64 + k];
        tl_all[((size_t)l * NBAT + b) * HD + h] = s;
    }
}

// ---- node update: agg = sum_j sigmoid(e_new)*Vh ; x += relu(LN(Uh+agg)) ---
__global__ __launch_bounds__(128) void k_node_update(const float* __restrict__ en_all,
        const float* __restrict__ Vh, const float* __restrict__ Uh,
        const float* __restrict__ g, const float* __restrict__ bta,
        float* __restrict__ x) {
    __shared__ float sb[128];
    int bi = blockIdx.x;                  // b*256 + i
    int h = threadIdx.x;
    int b = bi >> 8;
    const float* en = en_all + (size_t)bi * NPT * HD;
    const float* vh = Vh + ((size_t)(b << 8)) * HD;
    float agg = 0.f;
    for (int j = 0; j < NPT; ++j) {
        float gv = en[(size_t)j * HD + h];
        gv = 1.f / (1.f + __expf(-gv));
        agg += gv * vh[(size_t)j * HD + h];
    }
    float val = Uh[(size_t)bi * HD + h] + agg;
    float mu  = bred_sum(val, sb, h, HD) * (1.f / HD);
    float d   = val - mu;
    float var = bred_sum(d * d, sb, h, HD) * (1.f / HD);
    float y = d * rsqrtf(var + LN_EPS) * g[h] + bta[h];
    x[(size_t)bi * HD + h] += fmaxf(y, 0.f);
}

// ---- edge: e_new <- silu(LN2(relu(LN1(e_new)) + tl[b]))  (in place) -------
__global__ __launch_bounds__(128) void k_edge_ln(float* __restrict__ en,
        const float* __restrict__ lng, const float* __restrict__ lnb,
        const float* __restrict__ tl,  const float* __restrict__ pg,
        const float* __restrict__ pb) {
    __shared__ float sb[128];
    size_t m = blockIdx.x;
    int h = threadIdx.x;
    int b = (int)(m >> 16);
    float* row = en + m * HD;
    float v = row[h];
    float mu  = bred_sum(v, sb, h, HD) * (1.f / HD);
    float d   = v - mu;
    float var = bred_sum(d * d, sb, h, HD) * (1.f / HD);
    float y = d * rsqrtf(var + LN_EPS) * lng[h] + lnb[h];
    y = fmaxf(y, 0.f) + tl[(size_t)b * HD + h];
    float mu2  = bred_sum(y, sb, h, HD) * (1.f / HD);
    float d2   = y - mu2;
    float var2 = bred_sum(d2 * d2, sb, h, HD) * (1.f / HD);
    float z = d2 * rsqrtf(var2 + LN_EPS) * pg[h] + pb[h];
    row[h] = z / (1.f + __expf(-z));       // silu
}

// ---------------------------- group-norm stats -----------------------------
__global__ __launch_bounds__(256) void k_gn_stats(const float* __restrict__ e,
        float* __restrict__ mu, float* __restrict__ rs) {
    __shared__ float sb[256];
    int bg = blockIdx.x;                  // b*32 + g
    int b = bg >> 5, g = bg & 31;
    int t = threadIdx.x;
    float s = 0.f, s2 = 0.f;
    const int TOT = NPT * NPT * 4;
    for (int idx = t; idx < TOT; idx += 256) {
        int c = idx & 3;
        int ij = idx >> 2;
        float v = e[(((size_t)(b << 16) + ij) * HD) + (g << 2) + c];
        s += v; s2 += v * v;
    }
    float S  = bred_sum(s,  sb, t, 256);
    float S2 = bred_sum(s2, sb, t, 256);
    if (t == 0) {
        float m = S / (float)TOT;
        float var = S2 / (float)TOT - m * m;
        mu[bg] = m;
        rs[bg] = rsqrtf(var + LN_EPS);
    }
}

// ---------------------- head: GN affine + ReLU + 1x1 conv ------------------
__global__ void k_head(const float* __restrict__ e, const float* __restrict__ mu,
                       const float* __restrict__ rs, const float* __restrict__ gng,
                       const float* __restrict__ gnb, const float* __restrict__ cw,
                       const float* __restrict__ cb, float* __restrict__ out,
                       int total) {
    int m = blockIdx.x * blockDim.x + threadIdx.x;
    if (m >= total) return;
    int b = m >> 16;
    float acc = cb[0];
    const float* row = e + (size_t)m * HD;
    for (int h = 0; h < HD; ++h) {
        int gi = (b << 5) + (h >> 2);
        float v = (row[h] - mu[gi]) * rs[gi] * gng[h] + gnb[h];
        acc += fmaxf(v, 0.f) * cw[h];
    }
    out[m] = acc;
}

// ============================== launcher ===================================
extern "C" void kernel_launch(void* const* d_in, const int* in_sizes, int n_in,
                              void* d_out, int out_size, void* d_ws, size_t ws_size,
                              hipStream_t stream) {
    (void)in_sizes; (void)n_in; (void)out_size; (void)ws_size;
    auto F = [&](int i) { return (const float*)d_in[i]; };

    const float* cm = F(0);
    const int*   ts = (const int*)d_in[1];
    const int*   xt = (const int*)d_in[2];
    const float *nf_w1 = F(3), *nf_b1 = F(4), *nf_w2 = F(5), *nf_b2 = F(6);
    const float *ee_w = F(7), *ee_b = F(8);
    const float *t_w1 = F(9), *t_b1 = F(10), *t_w2 = F(11), *t_b2 = F(12);
    const float *gn_g = F(13), *gn_b = F(14), *conv_w = F(15), *conv_b = F(16);
    auto LP = [&](int l, int k) { return F(17 + l * 20 + k); };
    // per-layer: 0 Uw 1 Ub 2 Vw 3 Vb 4 Aw 5 Ab 6 Bw 7 Bb 8 Cw 9 Cb
    //            10 ln_h_g 11 ln_h_b 12 ln_e_g 13 ln_e_b 14 tl_w 15 tl_b
    //            16 plo_g 17 plo_b 18 plo_w 19 plo_bias

    // -------- workspace layout --------
    char* ws = (char*)d_ws;
    size_t off = 0;
    auto alloc = [&](size_t bytes) {
        void* p = ws + off;
        off += (bytes + 255) & ~(size_t)255;
        return p;
    };
    float*    E    = (float*)alloc((size_t)M_EDGE * HD * 4);   // 67 MB
    float*    EN   = (float*)alloc((size_t)M_EDGE * HD * 4);   // 67 MB
    _Float16* W16  = (_Float16*)alloc((size_t)19 * HD * HD * 2);
    float*    X    = (float*)alloc((size_t)M_NODE * HD * 4);
    float*    UH   = (float*)alloc((size_t)M_NODE * HD * 4);
    float*    VH   = (float*)alloc((size_t)M_NODE * HD * 4);
    float*    AH   = (float*)alloc((size_t)M_NODE * HD * 4);
    float*    BH   = (float*)alloc((size_t)M_NODE * HD * 4);
    float*    NF   = (float*)alloc((size_t)M_NODE * 8 * 4);
    float*    NH   = (float*)alloc((size_t)M_NODE * 256 * 4);
    float*    TL   = (float*)alloc((size_t)NLAYER * NBAT * HD * 4);
    float*    GNMU = (float*)alloc((size_t)NBAT * 32 * 4);
    float*    GNRS = (float*)alloc((size_t)NBAT * 32 * 4);

    auto Wslot = [&](int s) { return W16 + (size_t)s * HD * HD; };
    const int HH = HD * HD;

    // -------- convert all WMMA weights to f16 (stored [n][k] row-major) ----
    hipLaunchKernelGGL(k_cvt_f16, dim3(HH / 256), dim3(256), 0, stream, ee_w, Wslot(0), HH);
    for (int l = 0; l < NLAYER; ++l) {
        hipLaunchKernelGGL(k_cvt_f16, dim3(HH/256), dim3(256), 0, stream, LP(l,0),  Wslot(1+6*l+0), HH); // Uw
        hipLaunchKernelGGL(k_cvt_f16, dim3(HH/256), dim3(256), 0, stream, LP(l,2),  Wslot(1+6*l+1), HH); // Vw
        hipLaunchKernelGGL(k_cvt_f16, dim3(HH/256), dim3(256), 0, stream, LP(l,4),  Wslot(1+6*l+2), HH); // Aw
        hipLaunchKernelGGL(k_cvt_f16, dim3(HH/256), dim3(256), 0, stream, LP(l,6),  Wslot(1+6*l+3), HH); // Bw
        hipLaunchKernelGGL(k_cvt_f16, dim3(HH/256), dim3(256), 0, stream, LP(l,8),  Wslot(1+6*l+4), HH); // Cw
        hipLaunchKernelGGL(k_cvt_f16, dim3(HH/256), dim3(256), 0, stream, LP(l,18), Wslot(1+6*l+5), HH); // plo_w
    }

    // -------- node features / time embedding ------------------------------
    hipLaunchKernelGGL(k_node_stats, dim3(M_NODE), dim3(256), 0, stream, cm, NF);
    hipLaunchKernelGGL(k_nf1, dim3(M_NODE * 256 / 256), dim3(256), 0, stream, NF, nf_w1, nf_b1, NH);
    hipLaunchKernelGGL(k_nf2, dim3(M_NODE * HD / 256), dim3(256), 0, stream, NH, nf_w2, nf_b2, X);
    hipLaunchKernelGGL(k_temb, dim3(NBAT), dim3(128), 0, stream, ts, t_w1, t_b1, t_w2, t_b2,
                       LP(0,14), LP(0,15), LP(1,14), LP(1,15), LP(2,14), LP(2,15), TL);

    // -------- edge embedding (sine-fused WMMA GEMM) ------------------------
    const int EDGE_BLKS = M_EDGE / 16 / 8;   // 1024 blocks of 8 waves
    hipLaunchKernelGGL(k_gemm_sine, dim3(EDGE_BLKS), dim3(256), 0, stream,
                       cm, xt, Wslot(0), ee_b, E, M_EDGE);

    // -------- GNN layers ---------------------------------------------------
    const int NODE_BLKS = M_NODE / 16 / 8;   // 4 blocks
    for (int l = 0; l < NLAYER; ++l) {
        _Float16* Uw = Wslot(1+6*l+0); _Float16* Vw = Wslot(1+6*l+1);
        _Float16* Aw = Wslot(1+6*l+2); _Float16* Bw = Wslot(1+6*l+3);
        _Float16* Cw = Wslot(1+6*l+4); _Float16* Pw = Wslot(1+6*l+5);

        hipLaunchKernelGGL(k_gemm<0>, dim3(NODE_BLKS), dim3(256), 0, stream,
                           X, Uw, LP(l,1), nullptr, nullptr, nullptr, UH, M_NODE);
        hipLaunchKernelGGL(k_gemm<0>, dim3(NODE_BLKS), dim3(256), 0, stream,
                           X, Vw, LP(l,3), nullptr, nullptr, nullptr, VH, M_NODE);
        hipLaunchKernelGGL(k_gemm<0>, dim3(NODE_BLKS), dim3(256), 0, stream,
                           X, Aw, LP(l,5), nullptr, nullptr, nullptr, AH, M_NODE);
        hipLaunchKernelGGL(k_gemm<0>, dim3(NODE_BLKS), dim3(256), 0, stream,
                           X, Bw, LP(l,7), nullptr, nullptr, nullptr, BH, M_NODE);

        // e_new = e@Cw^T + Cb + Ah_j + Bh_i
        hipLaunchKernelGGL(k_gemm<1>, dim3(EDGE_BLKS), dim3(256), 0, stream,
                           E, Cw, LP(l,9), nullptr, AH, BH, EN, M_EDGE);

        // x += relu(LN(Uh + sum_j sigmoid(e_new)*Vh))
        hipLaunchKernelGGL(k_node_update, dim3(M_NODE), dim3(128), 0, stream,
                           EN, VH, UH, LP(l,10), LP(l,11), X);

        // e_new <- silu(LN2(relu(LN1(e_new)) + tl))
        hipLaunchKernelGGL(k_edge_ln, dim3(M_EDGE), dim3(128), 0, stream,
                           EN, LP(l,12), LP(l,13), TL + (size_t)l * NBAT * HD,
                           LP(l,16), LP(l,17));

        // e = e + e_new@plo_w^T + plo_bias   (in-place residual)
        hipLaunchKernelGGL(k_gemm<2>, dim3(EDGE_BLKS), dim3(256), 0, stream,
                           EN, Pw, LP(l,19), E, nullptr, nullptr, E, M_EDGE);
    }

    // -------- output head --------------------------------------------------
    hipLaunchKernelGGL(k_gn_stats, dim3(NBAT * 32), dim3(256), 0, stream, E, GNMU, GNRS);
    hipLaunchKernelGGL(k_head, dim3(M_EDGE / 256), dim3(256), 0, stream,
                       E, GNMU, GNRS, gn_g, gn_b, conv_w, conv_b, (float*)d_out, M_EDGE);
}
